// VectorQuantizer_49873160241296
// MI455X (gfx1250) — matrix operand order repaired and checked by reading, not compiled, not run on hardware
//
#include <hip/hip_runtime.h>
#include <stdint.h>

#define NUM_EMB 512
#define DIM 128
#define NTOK 524288
#define ROWS_PER_BLOCK 128
#define CHUNK_N 64
#define NCHUNK (NUM_EMB / CHUNK_N)
#define LDA 136  // padded bf16 row stride (272 B) -> no LDS bank conflicts on b128 reads

typedef __bf16 v16bf __attribute__((ext_vector_type(16)));
typedef float  v8f   __attribute__((ext_vector_type(8)));

union BF16Frag { v16bf v; uint4 q[2]; };

struct __align__(8) US4 { unsigned short a, b, c, d; };

__device__ __forceinline__ unsigned short f2bf_rne(float f) {
  unsigned int x = __float_as_uint(f);
  x += 0x7FFFu + ((x >> 16) & 1u);   // round-to-nearest-even to bf16
  return (unsigned short)(x >> 16);
}

// ---------------------------------------------------------------------------
// Prep: codebook -> bf16 in workspace, plus -0.5*||e_k||^2 in fp32.
// ---------------------------------------------------------------------------
__global__ void vq_prep(const float* __restrict__ emb,
                        unsigned short* __restrict__ embBf,
                        float* __restrict__ normNegHalf) {
  const int code = blockIdx.x;     // 512 blocks
  const int tid  = threadIdx.x;    // 128 threads
  float e = emb[code * DIM + tid];
  embBf[code * DIM + tid] = f2bf_rne(e);
  float s = e * e;
#pragma unroll
  for (int m = 1; m < 32; m <<= 1) s += __shfl_xor(s, m, 32);
  __shared__ float r4[4];
  if ((tid & 31) == 0) r4[tid >> 5] = s;
  __syncthreads();
  if (tid == 0) normNegHalf[code] = -0.5f * (r4[0] + r4[1] + r4[2] + r4[3]);
}

// ---------------------------------------------------------------------------
// Main: per block = 128 rows. argmax over k of (z.e - 0.5||e||^2) via bf16 WMMA,
// then exact fp32 gather + per-block partial loss sum.
// ---------------------------------------------------------------------------
__global__ __launch_bounds__(256) void vq_main(
    const float* __restrict__ z, const float* __restrict__ emb,
    const unsigned short* __restrict__ embBf,
    const float* __restrict__ normNegHalf,
    float* __restrict__ outQ, int* __restrict__ outIdx,
    float* __restrict__ partial) {

  __shared__ unsigned short As[ROWS_PER_BLOCK * LDA];  // 34816 B
  __shared__ unsigned short Bs[CHUNK_N * LDA];         // 17408 B
  __shared__ int   idxSm[ROWS_PER_BLOCK];
  __shared__ float redSm[8];

  const int tid    = threadIdx.x;
  const int lane   = tid & 31;
  const int wave   = tid >> 5;                 // 8 waves, one 16-row M tile each
  const int laneHi = (lane >> 4) & 1;          // 0: lanes 0-15, 1: lanes 16-31
  const int lane15 = lane & 15;
  const long long rowBase = (long long)blockIdx.x * ROWS_PER_BLOCK;

  // ---- Preload all 32 per-lane bias values (-0.5||e_n||^2) up front so the
  //      global-load latency hides behind Stage A instead of stalling each tile.
  float c0s[NCHUNK * 4];
#pragma unroll
  for (int chunk = 0; chunk < NCHUNK; ++chunk)
#pragma unroll
    for (int t = 0; t < 4; ++t)
      c0s[chunk * 4 + t] = normNegHalf[chunk * CHUNK_N + t * 16 + lane15];

  // ---- Stage A: z tile (128x128 f32) -> bf16 LDS. Batched: 8 loads in flight
  //      before the first conversion/store, twice.
#pragma unroll
  for (int half = 0; half < 2; ++half) {
    float4 tmp[8];
#pragma unroll
    for (int j = 0; j < 8; ++j) {
      int i4 = tid + (half * 8 + j) * 256;     // 4096 float4 groups total
      int m  = i4 >> 5;                        // 32 float4 per row
      int c  = i4 & 31;
      tmp[j] = *(const float4*)(z + (rowBase + m) * DIM + c * 4);
    }
#pragma unroll
    for (int j = 0; j < 8; ++j) {
      int i4 = tid + (half * 8 + j) * 256;
      int m  = i4 >> 5;
      int c  = i4 & 31;
      US4 u = { f2bf_rne(tmp[j].x), f2bf_rne(tmp[j].y),
                f2bf_rne(tmp[j].z), f2bf_rne(tmp[j].w) };
      *(US4*)(&As[m * LDA + c * 4]) = u;
    }
  }
  __syncthreads();

  // ---- A fragments for this wave's M tile: loaded once, reused for all 512 codes.
  // A 16x32 bf16 layout (ISA 7.12.2): lanes 0-15 K{0-7,16-23}, lanes 16-31 K{8-15,24-31}
  BF16Frag af[4];
  {
    const int row = wave * 16 + lane15;
#pragma unroll
    for (int s = 0; s < 4; ++s) {
      const int kOff = s * 32 + laneHi * 8;
      af[s].q[0] = *(const uint4*)(&As[row * LDA + kOff]);
      af[s].q[1] = *(const uint4*)(&As[row * LDA + kOff + 16]);
    }
  }

  float best[8];
  int   bidx[8];
#pragma unroll
  for (int r = 0; r < 8; ++r) { best[r] = -3.4e38f; bidx[r] = 0; }

  const uint64_t embBfBase = (uint64_t)(uintptr_t)embBf;

  for (int chunk = 0; chunk < NCHUNK; ++chunk) {
    __syncthreads();                           // previous chunk consumed Bs
    // ---- Stage B chunk (64 codes, already bf16 in workspace) straight into LDS
    //      via the CDNA5 async load-to-LDS path: no VGPR bounce, ASYNCcnt-tracked.
#pragma unroll
    for (int j = 0; j < 4; ++j) {
      int g = tid + j * 256;                   // 1024 x 16B = 16 KB per chunk
      int code = g >> 4;
      int part = g & 15;
      unsigned int voff  = (unsigned int)(((chunk * CHUNK_N + code) * DIM + part * 8) * 2);
      unsigned int laddr = (unsigned int)(uintptr_t)(&Bs[code * LDA + part * 8]);
      asm volatile("global_load_async_to_lds_b128 %0, %1, %2"
                   :: "v"(laddr), "v"(voff), "s"(embBfBase) : "memory");
    }
    asm volatile("s_wait_asynccnt 0x0" ::: "memory");
    __syncthreads();

#pragma unroll
    for (int t = 0; t < 4; ++t) {
      const int nl = t * 16 + lane15;          // code within chunk (= B column)
      const int n  = chunk * CHUNK_N + nl;     // global code index for this lane
      const float c0 = c0s[chunk * 4 + t];     // exact fp32 bias folded into C
      v8f acc;
#pragma unroll
      for (int r = 0; r < 8; ++r) acc[r] = c0;
#pragma unroll
      for (int s = 0; s < 4; ++s) {
        // B 32x16 bf16: lo lanes hold K{0-7,16-23}, hi lanes K{8-15,24-31}
        BF16Frag bf;
        const int kOff = s * 32 + laneHi * 8;
        bf.q[0] = *(const uint4*)(&Bs[nl * LDA + kOff]);
        bf.q[1] = *(const uint4*)(&Bs[nl * LDA + kOff + 16]);
        acc = __builtin_amdgcn_wmma_f32_16x16x32_bf16(
            false, af[s].v, false, bf.v, (short)0, acc, false, false);
      }
#pragma unroll
      for (int r = 0; r < 8; ++r) {
        if (acc[r] > best[r]) { best[r] = acc[r]; bidx[r] = n; }  // strict > keeps first min
      }
    }
  }

  // ---- Cross-lane argmax within each 16-lane half (C layout: column n = lane15)
#pragma unroll
  for (int r = 0; r < 8; ++r) {
#pragma unroll
    for (int m = 1; m < 16; m <<= 1) {
      float ov = __shfl_xor(best[r], m, 32);
      int   oi = __shfl_xor(bidx[r], m, 32);
      if (ov > best[r] || (ov == best[r] && oi < bidx[r])) { best[r] = ov; bidx[r] = oi; }
    }
  }
  if (lane15 == 0) {                           // lanes 0 (rows 0-7) and 16 (rows 8-15)
    const int mbase = wave * 16 + laneHi * 8;
#pragma unroll
    for (int r = 0; r < 8; ++r) {
      const int m = mbase + r;
      idxSm[m] = bidx[r];
      outIdx[rowBase + m] = bidx[r];
    }
  }
  __syncthreads();

  // ---- Gather exact fp32 codes + loss partial. z re-read is L2-hot.
  float lsum = 0.f;
#pragma unroll
  for (int j = 0; j < 16; ++j) {
    int i4 = tid + j * 256;
    int m  = i4 >> 5;
    int c  = i4 & 31;
    const int code = idxSm[m];
    float4 q  = *(const float4*)(emb + (size_t)code * DIM + c * 4);
    float4 zv = *(const float4*)(z + (rowBase + m) * DIM + c * 4);
    *(float4*)(outQ + (rowBase + m) * DIM + c * 4) = q;
    float dx = q.x - zv.x, dy = q.y - zv.y, dz = q.z - zv.z, dw = q.w - zv.w;
    lsum += dx * dx + dy * dy + dz * dz + dw * dw;
  }
#pragma unroll
  for (int m = 1; m < 32; m <<= 1) lsum += __shfl_xor(lsum, m, 32);
  if (lane == 0) redSm[wave] = lsum;
  __syncthreads();
  if (tid == 0) {
    float s = 0.f;
#pragma unroll
    for (int w = 0; w < 8; ++w) s += redSm[w];
    partial[blockIdx.x] = s;
  }
}

// ---------------------------------------------------------------------------
// Deterministic final reduction of 4096 block partials -> scalar loss.
// loss = q_latent + 0.25*e_latent = 1.25 * mean((q - z)^2)
// ---------------------------------------------------------------------------
__global__ void vq_finalize(const float* __restrict__ partial,
                            float* __restrict__ lossPtr) {
  __shared__ float sm[256];
  const int tid = threadIdx.x;
  float s = 0.f;
  for (int j = 0; j < 16; ++j) s += partial[tid * 16 + j];  // fixed order
  sm[tid] = s;
  __syncthreads();
  for (int off = 128; off > 0; off >>= 1) {
    if (tid < off) sm[tid] += sm[tid + off];
    __syncthreads();
  }
  if (tid == 0) *lossPtr = sm[0] * (1.25f / (float)((long long)NTOK * DIM));
}

extern "C" void kernel_launch(void* const* d_in, const int* in_sizes, int n_in,
                              void* d_out, int out_size, void* d_ws, size_t ws_size,
                              hipStream_t stream) {
  (void)in_sizes; (void)n_in; (void)out_size; (void)ws_size;
  const float* z   = (const float*)d_in[0];   // [524288, 128] f32
  const float* emb = (const float*)d_in[1];   // [512, 128] f32

  float* out      = (float*)d_out;
  float* outQ     = out;                                   // [N, D] f32
  float* lossPtr  = out + (size_t)NTOK * DIM;              // scalar
  int*   outIdx   = (int*)(out + (size_t)NTOK * DIM + 1);  // [N] int32 (bit-stored)

  unsigned short* embBf       = (unsigned short*)d_ws;                       // 128 KB
  float*          normNegHalf = (float*)((char*)d_ws + NUM_EMB * DIM * 2);   //   2 KB
  float*          partial     = normNegHalf + NUM_EMB;                       //  16 KB

  vq_prep<<<NUM_EMB, DIM, 0, stream>>>(emb, embBf, normNegHalf);
  vq_main<<<NTOK / ROWS_PER_BLOCK, 256, 0, stream>>>(z, emb, embBf, normNegHalf,
                                                     outQ, outIdx, partial);
  vq_finalize<<<1, 256, 0, stream>>>(partial, lossPtr);
}
